// GraphRNN_48086453846715
// MI455X (gfx1250) — matrix operand order repaired and checked by reading, not compiled, not run on hardware
//
#include <hip/hip_runtime.h>
#include <cmath>

#define TT     8
#define NN     20000
#define DD     64
#define HH     128
#define EE     640000
#define NGR    20
#define NATOMS 1000
#define NSS    8

typedef _Float16 half_t;
typedef __attribute__((ext_vector_type(16))) _Float16 v16h;
typedef __attribute__((ext_vector_type(8)))  _Float16 v8h;
typedef __attribute__((ext_vector_type(8)))  float    v8f;

__device__ __forceinline__ void atomAddF(float* p, float v) {
  // lowers to global_atomic_add_f32 (no CAS loop), relaxed, device scope
  __hip_atomic_fetch_add(p, v, __ATOMIC_RELAXED, __HIP_MEMORY_SCOPE_AGENT);
}

__global__ __launch_bounds__(256) void zero32(uint32_t* __restrict__ p, int n) {
  int i = blockIdx.x * 256 + threadIdx.x;
  if (i < n) p[i] = 0u;
}

__global__ __launch_bounds__(256) void degree_kernel(const int* __restrict__ dst,
                                                     int* __restrict__ deg, int ne) {
  int e = blockIdx.x * 256 + threadIdx.x;
  if (e < ne) atomicAdd(&deg[dst[e]], 1);
}

__global__ __launch_bounds__(256) void dis_kernel(const int* __restrict__ deg,
                                                  float* __restrict__ dis, int n) {
  int i = blockIdx.x * 256 + threadIdx.x;
  if (i < n) dis[i] = rsqrtf((float)(deg[i] + 1)); // +1 = self loop; always >= 1
}

__global__ __launch_bounds__(256) void cnt_kernel(const int* __restrict__ bid,
                                                  int* __restrict__ cnt, int n) {
  int i = blockIdx.x * 256 + threadIdx.x;
  if (i < n) atomicAdd(&cnt[bid[i]], 1);
}

// C[nrows x 128] = A[nrows x K] @ W[K x 128], f16 WMMA with f32 accumulation.
// Block: 256 threads (8 waves). 64 output rows per block, all 128 cols.
// Wave w: row tile rt = w&3, col tiles (w>>2)*4 .. +3.
__global__ __launch_bounds__(256) void gemm_wmma(const float* __restrict__ A,
                                                 const float* __restrict__ W,
                                                 float* __restrict__ C,
                                                 int nrows, int K) {
  extern __shared__ half_t lds[];
  const int ldA = K + 8;            // pad: 16B-aligned rows + bank-spread
  half_t* As = lds;                 // 64 x ldA
  half_t* Wt = lds + 64 * ldA;      // 128 x ldA  (W transposed: [n][k])

  const int tid = threadIdx.x;
  const int m_base = blockIdx.x * 64;

  for (int idx = tid; idx < 64 * K; idx += 256) {
    int r = idx / K, c = idx - r * K;
    int row = m_base + r;
    As[r * ldA + c] = (half_t)((row < nrows) ? A[(long)row * K + c] : 0.0f);
  }
  for (int idx = tid; idx < K * HH; idx += 256) {
    int k = idx >> 7, n = idx & 127;          // W row-major [K][128], coalesced read
    Wt[n * ldA + k] = (half_t)W[idx];
  }
  __syncthreads();

  const int wave = tid >> 5, lane = tid & 31;
  const int hi = lane >> 4, lm = lane & 15;
  const int rt = wave & 3;
  const int cbase = (wave >> 2) * 4;
  const half_t* arow = As + (rt * 16 + lm) * ldA;

  v8f acc[4] = {};
  for (int k0 = 0; k0 < K; k0 += 32) {
    // A fragment (16x32 f16): halves 0-7 -> K = k0+8*hi+h ; 8-15 -> K = k0+16+8*hi+h
    v8h alo = *(const v8h*)(arow + k0 + 8 * hi);
    v8h ahi = *(const v8h*)(arow + k0 + 16 + 8 * hi);
    v16h af = __builtin_shufflevector(alo, ahi, 0,1,2,3,4,5,6,7,8,9,10,11,12,13,14,15);
#pragma unroll
    for (int cc = 0; cc < 4; ++cc) {
      // B fragment (32x16): N = lm, halves 0-7 -> K = k0+8*hi+h ; 8-15 -> K = k0+16+8*hi+h
      const half_t* bcol = Wt + ((cbase + cc) * 16 + lm) * ldA + k0 + 8 * hi;
      v8h blo = *(const v8h*)(bcol);
      v8h bhi = *(const v8h*)(bcol + 16);
      v16h bf = __builtin_shufflevector(blo, bhi, 0,1,2,3,4,5,6,7,8,9,10,11,12,13,14,15);
      acc[cc] = __builtin_amdgcn_wmma_f32_16x16x32_f16(
          false, af, false, bf, (short)0, acc[cc], false, false);
    }
  }
#pragma unroll
  for (int cc = 0; cc < 4; ++cc) {
    int col = (cbase + cc) * 16 + lm;
#pragma unroll
    for (int r = 0; r < 8; ++r) {
      int row = m_base + rt * 16 + hi * 8 + r;   // D vgpr r: M = r + 8*hi
      if (row < nrows) C[(long)row * HH + col] = acc[cc][r];
    }
  }
}

// agg[i][c] = m[i][c] * dis[i]^2   (self-loop term initializes the accumulator)
__global__ __launch_bounds__(256) void self_init(const float* __restrict__ m,
                                                 const float* __restrict__ dis,
                                                 float* __restrict__ agg) {
  int i = blockIdx.x * 256 + threadIdx.x;
  if (i >= NN * HH) return;
  float d = dis[i >> 7];
  agg[i] = m[i] * d * d;
}

// one wave per edge: agg[dst] += m[src] * dis[src]*dis[dst]  (128 ch, 4/lane)
__global__ __launch_bounds__(256) void edge_scatter(const float* __restrict__ m,
                                                    const int* __restrict__ src,
                                                    const int* __restrict__ dst,
                                                    const float* __restrict__ dis,
                                                    float* __restrict__ agg, int ne) {
  int gw = (blockIdx.x * 256 + threadIdx.x) >> 5;
  int lane = threadIdx.x & 31;
  int nw = (gridDim.x * 256) >> 5;
  for (int e = gw; e < ne; e += nw) {
    int s = src[e], d = dst[e];
    float w = dis[s] * dis[d];
    float4 mv = ((const float4*)(m + (long)s * HH))[lane];
    float* ap = agg + (long)d * HH + lane * 4;
    atomAddF(ap + 0, mv.x * w);
    atomAddF(ap + 1, mv.y * w);
    atomAddF(ap + 2, mv.z * w);
    atomAddF(ap + 3, mv.w * w);
  }
}

__global__ __launch_bounds__(256) void bias_relu(const float* __restrict__ agg,
                                                 const float* __restrict__ b,
                                                 float* __restrict__ h) {
  int i = blockIdx.x * 256 + threadIdx.x;
  if (i >= NN * HH) return;
  float v = agg[i] + b[i & 127];
  h[i] = v > 0.0f ? v : 0.0f;
}

__global__ __launch_bounds__(256) void pool_acc(const float* __restrict__ h,
                                                const int* __restrict__ bid,
                                                float* __restrict__ seqt) {
  int i = blockIdx.x * 256 + threadIdx.x;
  if (i >= NN * HH) return;
  atomAddF(&seqt[bid[i >> 7] * HH + (i & 127)], h[i]);
}

// Single-block GRU over T steps. B=20, H=128 -> 2560 state elements.
__global__ __launch_bounds__(256) void gru_kernel(const float* __restrict__ seq,
                                                  const int* __restrict__ cnt,
                                                  const float* __restrict__ wih,
                                                  const float* __restrict__ whh,
                                                  const float* __restrict__ bih,
                                                  const float* __restrict__ bhh,
                                                  float* __restrict__ outs) {
  __shared__ float xt[NGR * HH];
  __shared__ float hs[NGR * HH];
  __shared__ float inv[NGR];
  const int tid = threadIdx.x;
  for (int i = tid; i < NGR * HH; i += 256) hs[i] = 0.0f;
  if (tid < NGR) { int c = cnt[tid]; inv[tid] = 1.0f / (float)(c > 1 ? c : 1); }
  __syncthreads();
  for (int t = 0; t < TT; ++t) {
    for (int i = tid; i < NGR * HH; i += 256) xt[i] = seq[t * NGR * HH + i] * inv[i >> 7];
    __syncthreads();
    float hnew[10];
    int q = 0;
    for (int i = tid; i < NGR * HH; i += 256, ++q) {
      int g = i >> 7, j = i & 127;
      const float* xg = xt + g * HH;
      const float* hg = hs + g * HH;
      float ir = bih[j], iz = bih[HH + j], in = bih[2 * HH + j];
      float hr = bhh[j], hz = bhh[HH + j], hn = bhh[2 * HH + j];
      const float* wr = wih + (long)j * HH;
      const float* wz = wih + (long)(HH + j) * HH;
      const float* wn = wih + (long)(2 * HH + j) * HH;
      const float* vr = whh + (long)j * HH;
      const float* vz = whh + (long)(HH + j) * HH;
      const float* vn = whh + (long)(2 * HH + j) * HH;
      for (int k = 0; k < HH; ++k) {
        float xk = xg[k], hk = hg[k];
        ir += xk * wr[k]; iz += xk * wz[k]; in += xk * wn[k];
        hr += hk * vr[k]; hz += hk * vz[k]; hn += hk * vn[k];
      }
      float r = 1.0f / (1.0f + expf(-(ir + hr)));
      float z = 1.0f / (1.0f + expf(-(iz + hz)));
      float nc = tanhf(in + r * hn);
      hnew[q] = (1.0f - z) * nc + z * hg[j];
    }
    __syncthreads();
    q = 0;
    for (int i = tid; i < NGR * HH; i += 256, ++q) {
      hs[i] = hnew[q];
      outs[t * NGR * HH + i] = hnew[q];
    }
    __syncthreads();
  }
}

// out = concat(coord[B,T,3000], ss[B,T,8000]); one thread per output element.
__global__ __launch_bounds__(256) void heads_kernel(const float* __restrict__ outs,
                                                    const float* __restrict__ cw,
                                                    const float* __restrict__ cb,
                                                    const float* __restrict__ sw,
                                                    const float* __restrict__ sb,
                                                    float* __restrict__ out) {
  const int CSZ = NGR * TT * NATOMS * 3;
  const int SSZ = NGR * TT * NATOMS * NSS;
  int idx = blockIdx.x * 256 + threadIdx.x;
  if (idx >= CSZ + SSZ) return;
  const float* act; const float* wrow; float bias;
  if (idx < CSZ) {
    int j = idx % (NATOMS * 3);
    int bt = idx / (NATOMS * 3);
    int t = bt % TT, b = bt / TT;
    act = outs + ((long)t * NGR + b) * HH;
    wrow = cw + (long)j * HH;
    bias = cb[j];
  } else {
    int k = idx - CSZ;
    int j = k % (NATOMS * NSS);
    int bt = k / (NATOMS * NSS);
    int t = bt % TT, b = bt / TT;
    act = outs + ((long)t * NGR + b) * HH;
    wrow = sw + (long)j * HH;
    bias = sb[j];
  }
  float acc = bias;
  const float4* a4 = (const float4*)act;
  const float4* w4 = (const float4*)wrow;
#pragma unroll 8
  for (int k4 = 0; k4 < HH / 4; ++k4) {
    float4 a = a4[k4], w = w4[k4];
    acc += a.x * w.x + a.y * w.y + a.z * w.z + a.w * w.w;
  }
  out[idx] = acc;
}

extern "C" void kernel_launch(void* const* d_in, const int* in_sizes, int n_in,
                              void* d_out, int out_size, void* d_ws, size_t ws_size,
                              hipStream_t stream) {
  const float* x   = (const float*)d_in[0];
  const int*   ei  = (const int*)d_in[1];
  const int*   bid = (const int*)d_in[2];
  const float* w0  = (const float*)d_in[3];
  const float* b0  = (const float*)d_in[4];
  const float* gw  = (const float*)d_in[5];
  const float* gb  = (const float*)d_in[6];
  const float* wih = (const float*)d_in[7];
  const float* whh = (const float*)d_in[8];
  const float* bih = (const float*)d_in[9];
  const float* bhh = (const float*)d_in[10];
  const float* cw  = (const float*)d_in[11];
  const float* cb  = (const float*)d_in[12];
  const float* sw  = (const float*)d_in[13];
  const float* sb  = (const float*)d_in[14];
  float* out = (float*)d_out;

  // workspace layout (~31 MB)
  float* hbuf = (float*)d_ws;
  float* mbuf = hbuf + (long)NN * HH;
  float* abuf = mbuf + (long)NN * HH;
  float* dis  = abuf + (long)NN * HH;
  int*   deg  = (int*)(dis + NN);
  int*   cnt  = deg + NN;
  float* seq  = (float*)(cnt + NGR);
  float* outs = seq + TT * NGR * HH;

  const int* src = ei;
  const int* dst = ei + EE;

  // degree / normalization / segment counts (deterministic, recomputed each call)
  zero32<<<(NN + 255) / 256, 256, 0, stream>>>((uint32_t*)deg, NN);
  zero32<<<1, 256, 0, stream>>>((uint32_t*)cnt, NGR);
  zero32<<<(TT * NGR * HH + 255) / 256, 256, 0, stream>>>((uint32_t*)seq, TT * NGR * HH);
  degree_kernel<<<(EE + 255) / 256, 256, 0, stream>>>(dst, deg, EE);
  dis_kernel<<<(NN + 255) / 256, 256, 0, stream>>>(deg, dis, NN);
  cnt_kernel<<<(NN + 255) / 256, 256, 0, stream>>>(bid, cnt, NN);

  const int ew_blocks = (NN * HH + 255) / 256;

  for (int t = 0; t < TT; ++t) {
    const float* in0 = x + (long)t * NN * DD;
    for (int layer = 0; layer < 4; ++layer) {
      const float* Ain  = (layer == 0) ? in0 : hbuf;
      const int    K    = (layer == 0) ? DD : HH;
      const float* W    = (layer == 0) ? w0 : gw + (long)(layer - 1) * HH * HH;
      const float* bias = (layer == 0) ? b0 : gb + (layer - 1) * HH;
      size_t ldsB = (size_t)(64 + 128) * (K + 8) * sizeof(half_t);  // <= 52 KB
      gemm_wmma<<<(NN + 63) / 64, 256, ldsB, stream>>>(Ain, W, mbuf, NN, K);
      self_init<<<ew_blocks, 256, 0, stream>>>(mbuf, dis, abuf);
      edge_scatter<<<2048, 256, 0, stream>>>(mbuf, src, dst, dis, abuf, EE);
      bias_relu<<<ew_blocks, 256, 0, stream>>>(abuf, bias, hbuf);
    }
    pool_acc<<<ew_blocks, 256, 0, stream>>>(hbuf, bid, seq + (long)t * NGR * HH);
  }

  gru_kernel<<<1, 256, 0, stream>>>(seq, cnt, wih, whh, bih, bhh, outs);

  const int total = NGR * TT * NATOMS * 3 + NGR * TT * NATOMS * NSS;  // 1,760,000
  heads_kernel<<<(total + 255) / 256, 256, 0, stream>>>(outs, cw, cb, sw, sb, out);
}
